// ForwardDeformer_43696997270175
// MI455X (gfx1250) — matrix operand order repaired and checked by reading, not compiled, not run on hardware
//
#include <hip/hip_runtime.h>
#include <hip/hip_bf16.h>
#include <math.h>
#include <stdint.h>

typedef __attribute__((ext_vector_type(16))) _Float16 v16h;
typedef __attribute__((ext_vector_type(8)))  float    v8f;
typedef const _Float16 __attribute__((address_space(1))) GHalf;  // global f16

#define B_   2
#define N_   2048
#define J_   24
#define NI_  9
#define CD_  16
#define PI_  (B_*N_*NI_)          // 36864
#define MAX_STEPS_  50
#define CVG_        1e-5f

#define LOG2E_      1.4426950408889634f
#define LN2_        0.6931471805599453f
#define SB_LOG2E_   28.853900817779268f   // 20 * log2(e)

// f16 fragment-packed weight arena in d_ws (halfs):
//  lbs : L0 @0 (KB=1,NT=8)  L1 @4096  L2 @20480  L3 @36864 (KB=4,NT=2)
//  disp: L0 @40960          L1 @45056 L2 @61440  L3 @77824 (KB=4,NT=1)
#define WS_HALFS     79872
#define NET_DISP_OFF 40960

// Dynamic-LDS arena byte offsets (one extern __shared__ symbol so the
// noinline MLP body sees addrspace(3) directly -> pure ds_load/ds_store).
#define SA_OFF_B     0        // A-staging: 4 waves * 2 mt * 4 kb * 512 halfs = 32KB
#define SOUT_OFF_B   32768    // layer-3 out: 4 waves * 32 rows * 32 f32 = 16KB
#define ST_OFF_B     49152    // bone transforms: 24 * 12 f32
#define SBIAS_OFF_B  50304    // biases: 795 f32
#define SMEM_BYTES   53504

// Bias float-index layout: lbs b0@0 b1@128 b2@256 b3@384(24); disp @408...
#define BIAS_DISP_OFF 408
#define BIAS_TOTAL    795

extern __shared__ __align__(32) char fd_smem[];

__device__ const int c_init_bones[NI_] = {0, 1, 2, 4, 5, 16, 17, 18, 19};

// ---------------------------------------------------------------------------
// Prep: pack fp32 row-major [K,N] weights into f16 WMMA B-fragment layout.
// Fragment slot = (nt*KB + kb); element = slot*512 + lane*16 + half;
// B layout: N = 16*nt + (lane&15); K = 32*kb + 8*(lane>>4) + half + 8*(half>>3).
// ---------------------------------------------------------------------------
__global__ void fd_prep_weights(_Float16* __restrict__ ws,
    const float* lw0, const float* lw1, const float* lw2, const float* lw3,
    const float* dw0, const float* dw1, const float* dw2, const float* dw3)
{
    int idx = blockIdx.x * blockDim.x + threadIdx.x;
    if (idx >= WS_HALFS) return;

    int off, KB, rK, rN; const float* W;
    if      (idx < 4096)  { off = 0;     KB = 1; rK = 19;  rN = 128; W = lw0; }
    else if (idx < 20480) { off = 4096;  KB = 4; rK = 128; rN = 128; W = lw1; }
    else if (idx < 36864) { off = 20480; KB = 4; rK = 128; rN = 128; W = lw2; }
    else if (idx < 40960) { off = 36864; KB = 4; rK = 128; rN = 24;  W = lw3; }
    else if (idx < 45056) { off = 40960; KB = 1; rK = 19;  rN = 128; W = dw0; }
    else if (idx < 61440) { off = 45056; KB = 4; rK = 128; rN = 128; W = dw1; }
    else if (idx < 77824) { off = 61440; KB = 4; rK = 128; rN = 128; W = dw2; }
    else                  { off = 77824; KB = 4; rK = 128; rN = 3;   W = dw3; }

    int e    = idx - off;
    int slot = e >> 9;
    int lane = (e >> 4) & 31;
    int half = e & 15;
    int nt   = slot / KB;
    int kb   = slot % KB;
    int K    = 32*kb + 8*(lane >> 4) + half + ((half >= 8) ? 8 : 0);
    int Ncol = 16*nt + (lane & 15);
    float v  = (K < rK && Ncol < rN) ? W[K*rN + Ncol] : 0.f;
    ws[idx]  = (_Float16)v;
}

// ---------------------------------------------------------------------------
// softplus with bare hardware v_exp_f32 / v_log_f32 (exp2/log2 semantics).
// Argument of the log is 1+e in [1,2] -> never subnormal, no fixup needed.
__device__ __forceinline__ float softplus_f(float x) {
    float e = __builtin_amdgcn_exp2f(-LOG2E_ * fabsf(x));
    return fmaxf(x, 0.f) + LN2_ * __builtin_amdgcn_logf(1.f + e);
}

__device__ __forceinline__ void inv3x3(const float* A, float* O) {
    float c00 = A[4]*A[8] - A[5]*A[7];
    float c01 = A[5]*A[6] - A[3]*A[8];
    float c02 = A[3]*A[7] - A[4]*A[6];
    float det = A[0]*c00 + A[1]*c01 + A[2]*c02;
    float id  = 1.f / det;
    O[0] = c00*id; O[1] = (A[2]*A[7]-A[1]*A[8])*id; O[2] = (A[1]*A[5]-A[2]*A[4])*id;
    O[3] = c01*id; O[4] = (A[0]*A[8]-A[2]*A[6])*id; O[5] = (A[2]*A[3]-A[0]*A[5])*id;
    O[6] = c02*id; O[7] = (A[1]*A[6]-A[0]*A[7])*id; O[8] = (A[0]*A[4]-A[1]*A[3])*id;
}

// Wave-local 4-layer MLP: 32 rows (2 M-tiles) per wave, all layers via
// v_wmma_f32_16x16x32_f16.  A fragments streamed from this wave's LDS
// staging region; outputs of layers 0..2 scattered back (softplus'd, f16)
// into A-fragment layout; layer-3 written f32 to sOut[row*32 + col].
// noinline keeps one code copy; LDS is reached through the module-scope
// dynamic-shared symbol (addrspace(3) inferred), weights through an
// explicit addrspace(1) pointer (global_load, not flat).
__device__ __attribute__((noinline)) void mlp_eval(
    GHalf* wnet,          // global, fragment-packed, at this net's base
    int biasBase,         // float index into LDS bias arena
    int sAwBase,          // half index of this wave's A staging
    int outBase,          // float index of this wave's 32x32 out tile
    int lane, int nt3, int rN3)
{
    _Float16* sA    = (_Float16*)(fd_smem + SA_OFF_B);
    float*    sOut  = (float*)(fd_smem + SOUT_OFF_B);
    float*    sBias = (float*)(fd_smem + SBIAS_OFF_B);

    const int m  = lane & 15;
    const int hi = lane >> 4;
    const int layOff [4] = {0, 4096, 20480, 36864};
    const int layKB  [4] = {1, 4, 4, 4};
    const int biasOff[4] = {0, 128, 256, 384};

    #pragma unroll
    for (int l = 0; l < 4; ++l) {
        const int KB = layKB[l];
        const int NT = (l < 3) ? 8   : nt3;
        const int rN = (l < 3) ? 128 : rN3;
        GHalf* wl = wnet + layOff[l];
        const int bb = biasBase + biasOff[l];

        for (int mt = 0; mt < 2; ++mt) {
            // Pre-load ALL A fragments for this M-tile before any scatter
            // stores overwrite the staging region (single-buffer reuse).
            v16h A[4];
            #pragma unroll
            for (int kb = 0; kb < 4; ++kb)
                if (kb < KB)
                    A[kb] = *(const v16h*)(sA + sAwBase + (mt*4 + kb)*512 + lane*16);

            #pragma unroll 2
            for (int nt = 0; nt < NT; ++nt) {
                v8f acc = {};
                #pragma unroll
                for (int kb = 0; kb < 4; ++kb) {
                    if (kb < KB) {
                        v16h Bf = *(const v16h*)(wl + (nt*KB + kb)*512 + lane*16);
                        acc = __builtin_amdgcn_wmma_f32_16x16x32_f16(
                                  false, A[kb], false, Bf,
                                  (short)0, acc, false, false);
                    }
                }
                const int   ncol = nt*16 + m;
                const float bv   = (ncol < rN) ? sBias[bb + ncol] : 0.f;
                #pragma unroll
                for (int r = 0; r < 8; ++r) {
                    const int row = r + 8*hi;          // C layout: M = r + 8*(lane>=16)
                    float v = acc[r] + bv;
                    if (l < 3) {
                        v = softplus_f(v);
                        const int k    = ncol;          // output col -> next-layer K
                        const int kb2  = k >> 5;
                        const int kk16 = k & 15;
                        const int lt   = (kk16 < 8) ? row : (row + 16);
                        const int h    = (((k & 31) >> 4) << 3) + (kk16 & 7);
                        sA[sAwBase + (mt*4 + kb2)*512 + lt*16 + h] = (_Float16)v;
                    } else {
                        sOut[outBase + (row + 16*mt)*32 + ncol] = v;
                    }
                }
            }
        }
    }
}

// ---------------------------------------------------------------------------
__launch_bounds__(128, 2)
__global__ void fd_deformer_kernel(
    const _Float16* __restrict__ ws,
    const float* __restrict__ xd, const float* __restrict__ cond,
    const float* __restrict__ tfs,
    const float* lb0, const float* lb1, const float* lb2, const float* lb3,
    const float* db0, const float* db1, const float* db2, const float* db3,
    float* __restrict__ out)
{
    _Float16* sA    = (_Float16*)(fd_smem + SA_OFF_B);
    float*    sOut  = (float*)(fd_smem + SOUT_OFF_B);
    float*    sT    = (float*)(fd_smem + ST_OFF_B);
    float*    sBias = (float*)(fd_smem + SBIAS_OFF_B);

    const int tid  = threadIdx.x;
    const int wave = tid >> 5;
    const int lane = tid & 31;
    const int p    = blockIdx.x * 128 + tid;          // PI index, grid exact
    const int b    = p / (N_*NI_);
    const int rem  = p % (N_*NI_);
    const int n    = rem / NI_;
    const int ib   = rem % NI_;

    GHalf* wsg = (GHalf*)(uintptr_t)(const void*)ws;  // global addrspace view

    // Stage this batch's bone transforms (blocks never straddle b).
    for (int idx = tid; idx < J_*12; idx += 128) {
        int j = idx / 12, c = idx % 12;
        float v = (c < 9) ? tfs[(size_t)(b*J_ + j)*16 + (c/3)*4 + (c%3)]
                          : tfs[(size_t)(b*J_ + j)*16 + (c-9)*4 + 3];
        sT[idx] = v;
    }
    // Stage all biases into LDS.
    for (int idx = tid; idx < BIAS_TOTAL; idx += 128) {
        float v;
        int t = idx;
        if      (t < 128) v = lb0[t];
        else if (t < 256) v = lb1[t - 128];
        else if (t < 384) v = lb2[t - 256];
        else if (t < 408) v = lb3[t - 384];
        else if (t < 536) v = db0[t - 408];
        else if (t < 664) v = db1[t - 536];
        else if (t < 792) v = db2[t - 664];
        else              v = db3[t - 792];
        sBias[idx] = v;
    }
    __syncthreads();

    __builtin_prefetch(ws, 0, 1);                     // global_prefetch_b8
    __builtin_prefetch(ws + 32768, 0, 1);

    const int sAwBase = wave * (2*4*512);             // half index
    const int outBase = wave * (32*32);               // float index
    const int m  = lane & 15;
    const int mt = lane >> 4;

    float xdv[3];
    #pragma unroll
    for (int d = 0; d < 3; ++d) xdv[d] = xd[(size_t)(b*N_ + n)*3 + d];

    // Layer-0 input row [x(3) | cond(16) | 0-pad] pre-packed into the two
    // 16-half lane vectors of the A-fragment layout; only halfs 0..2 of rowA
    // (K=0..2) change per evaluation.
    float in32[32];
    #pragma unroll
    for (int k = 0; k < 32; ++k) in32[k] = 0.f;
    #pragma unroll
    for (int j = 0; j < CD_; ++j) in32[3 + j] = cond[b*CD_ + j];
    v16h rowA, rowB;
    #pragma unroll
    for (int h = 0; h < 16; ++h) {
        rowA[h] = (_Float16)((h < 8) ? in32[h]     : in32[8 + h]);   // K 0-7,16-23
        rowB[h] = (_Float16)((h < 8) ? in32[8 + h] : in32[16 + h]);  // K 8-15,24-31
    }

    // g(x) = blend(softmax(20*MLP([x,cond])), T, x) - xd ; f = blend part.
    auto evalF = [&](float x0v, float x1v, float x2v, float f[3]) {
        rowA[0] = (_Float16)x0v; rowA[1] = (_Float16)x1v; rowA[2] = (_Float16)x2v;
        *(v16h*)(sA + sAwBase + mt*4*512 +  m      *16) = rowA;
        *(v16h*)(sA + sAwBase + mt*4*512 + (m + 16)*16) = rowB;
        mlp_eval(wsg, 0, sAwBase, outBase, lane, /*nt3=*/2, /*rN3=*/24);
        const float* lg = sOut + outBase + lane*32;   // staging row == lane
        float mx = lg[0];
        for (int j = 1; j < J_; ++j) mx = fmaxf(mx, lg[j]);
        float s = 0.f, a0 = 0.f, a1 = 0.f, a2 = 0.f;
        for (int j = 0; j < J_; ++j) {
            float e = __builtin_amdgcn_exp2f(SB_LOG2E_ * (lg[j] - mx));
            const float* Tr = sT + j*12;
            s  += e;
            a0 += e * (Tr[0]*x0v + Tr[1]*x1v + Tr[2]*x2v + Tr[9]);
            a1 += e * (Tr[3]*x0v + Tr[4]*x1v + Tr[5]*x2v + Tr[10]);
            a2 += e * (Tr[6]*x0v + Tr[7]*x1v + Tr[8]*x2v + Tr[11]);
        }
        float is = 1.f / s;
        f[0] = a0*is; f[1] = a1*is; f[2] = a2*is;
    };

    // xc_init = R^{-1} (xd - t) for this point's init bone.
    float T0[12];
    {
        const float* Tp = sT + c_init_bones[ib]*12;
        #pragma unroll
        for (int c = 0; c < 12; ++c) T0[c] = Tp[c];
    }
    float Ri[9];
    inv3x3(T0, Ri);
    float d0 = xdv[0]-T0[9], d1 = xdv[1]-T0[10], d2 = xdv[2]-T0[11];
    float x[3] = { Ri[0]*d0 + Ri[1]*d1 + Ri[2]*d2,
                   Ri[3]*d0 + Ri[4]*d1 + Ri[5]*d2,
                   Ri[6]*d0 + Ri[7]*d1 + Ri[8]*d2 };

    // Broyden seed Jacobian via forward differences (3 extra evals).
    float f0[3], fp[3], Jm[9];
    evalF(x[0], x[1], x[2], f0);
    const float eps = 1e-2f, iev = 1e2f;
    evalF(x[0]+eps, x[1], x[2], fp);
    Jm[0] = (fp[0]-f0[0])*iev; Jm[3] = (fp[1]-f0[1])*iev; Jm[6] = (fp[2]-f0[2])*iev;
    evalF(x[0], x[1]+eps, x[2], fp);
    Jm[1] = (fp[0]-f0[0])*iev; Jm[4] = (fp[1]-f0[1])*iev; Jm[7] = (fp[2]-f0[2])*iev;
    evalF(x[0], x[1], x[2]+eps, fp);
    Jm[2] = (fp[0]-f0[0])*iev; Jm[5] = (fp[1]-f0[1])*iev; Jm[8] = (fp[2]-f0[2])*iev;
    float Ji[9];
    inv3x3(Jm, Ji);

    float gx[3] = { f0[0]-xdv[0], f0[1]-xdv[1], f0[2]-xdv[2] };
    float upd[3] = { -(Ji[0]*gx[0] + Ji[1]*gx[1] + Ji[2]*gx[2]),
                     -(Ji[3]*gx[0] + Ji[4]*gx[1] + Ji[5]*gx[2]),
                     -(Ji[6]*gx[0] + Ji[7]*gx[1] + Ji[8]*gx[2]) };
    float bx[3] = { x[0], x[1], x[2] };
    float berr  = sqrtf(gx[0]*gx[0] + gx[1]*gx[1] + gx[2]*gx[2]);

    for (int it = 0; it < MAX_STEPS_; ++it) {
        float dx[3] = { upd[0], upd[1], upd[2] };
        x[0] += dx[0]; x[1] += dx[1]; x[2] += dx[2];
        float fn[3];
        evalF(x[0], x[1], x[2], fn);
        float gn[3] = { fn[0]-xdv[0], fn[1]-xdv[1], fn[2]-xdv[2] };
        float dg[3] = { gn[0]-gx[0], gn[1]-gx[1], gn[2]-gx[2] };
        float v0 = dx[0]*Ji[0] + dx[1]*Ji[3] + dx[2]*Ji[6];
        float v1 = dx[0]*Ji[1] + dx[1]*Ji[4] + dx[2]*Ji[7];
        float v2 = dx[0]*Ji[2] + dx[1]*Ji[5] + dx[2]*Ji[8];
        float a0 = dx[0] - (Ji[0]*dg[0] + Ji[1]*dg[1] + Ji[2]*dg[2]);
        float a1 = dx[1] - (Ji[3]*dg[0] + Ji[4]*dg[1] + Ji[5]*dg[2]);
        float a2 = dx[2] - (Ji[6]*dg[0] + Ji[7]*dg[1] + Ji[8]*dg[2]);
        float ibd = 1.f / (v0*dg[0] + v1*dg[1] + v2*dg[2] + 1e-6f);
        Ji[0] += a0*v0*ibd; Ji[1] += a0*v1*ibd; Ji[2] += a0*v2*ibd;
        Ji[3] += a1*v0*ibd; Ji[4] += a1*v1*ibd; Ji[5] += a1*v2*ibd;
        Ji[6] += a2*v0*ibd; Ji[7] += a2*v1*ibd; Ji[8] += a2*v2*ibd;
        upd[0] = -(Ji[0]*gn[0] + Ji[1]*gn[1] + Ji[2]*gn[2]);
        upd[1] = -(Ji[3]*gn[0] + Ji[4]*gn[1] + Ji[5]*gn[2]);
        upd[2] = -(Ji[6]*gn[0] + Ji[7]*gn[1] + Ji[8]*gn[2]);
        gx[0] = gn[0]; gx[1] = gn[1]; gx[2] = gn[2];
        float err = sqrtf(gn[0]*gn[0] + gn[1]*gn[1] + gn[2]*gn[2]);
        if (err < berr) { berr = err; bx[0] = x[0]; bx[1] = x[1]; bx[2] = x[2]; }
    }

    // corr term is identically zero in value semantics; go straight to the
    // displacement network: xc = bx + disp_mlp([bx, cond]).
    rowA[0] = (_Float16)bx[0]; rowA[1] = (_Float16)bx[1]; rowA[2] = (_Float16)bx[2];
    *(v16h*)(sA + sAwBase + mt*4*512 +  m      *16) = rowA;
    *(v16h*)(sA + sAwBase + mt*4*512 + (m + 16)*16) = rowB;
    mlp_eval(wsg + NET_DISP_OFF, BIAS_DISP_OFF, sAwBase, outBase, lane,
             /*nt3=*/1, /*rN3=*/3);

    out[(size_t)p*3 + 0] = bx[0] + sOut[outBase + lane*32 + 0];
    out[(size_t)p*3 + 1] = bx[1] + sOut[outBase + lane*32 + 1];
    out[(size_t)p*3 + 2] = bx[2] + sOut[outBase + lane*32 + 2];
    out[(size_t)(PI_*3) + p] = (berr < CVG_) ? 1.f : 0.f;
}

// ---------------------------------------------------------------------------
extern "C" void kernel_launch(void* const* d_in, const int* in_sizes, int n_in,
                              void* d_out, int out_size, void* d_ws, size_t ws_size,
                              hipStream_t stream)
{
    const float* xd   = (const float*)d_in[0];
    const float* cond = (const float*)d_in[1];
    const float* tfs  = (const float*)d_in[2];
    const float* lw0  = (const float*)d_in[3];
    const float* lb0  = (const float*)d_in[4];
    const float* lw1  = (const float*)d_in[5];
    const float* lb1  = (const float*)d_in[6];
    const float* lw2  = (const float*)d_in[7];
    const float* lb2  = (const float*)d_in[8];
    const float* lw3  = (const float*)d_in[9];
    const float* lb3  = (const float*)d_in[10];
    const float* dw0  = (const float*)d_in[11];
    const float* db0  = (const float*)d_in[12];
    const float* dw1  = (const float*)d_in[13];
    const float* db1  = (const float*)d_in[14];
    const float* dw2  = (const float*)d_in[15];
    const float* db2  = (const float*)d_in[16];
    const float* dw3  = (const float*)d_in[17];
    const float* db3  = (const float*)d_in[18];

    _Float16* ws = (_Float16*)d_ws;

    fd_prep_weights<<<(WS_HALFS + 255)/256, 256, 0, stream>>>(
        ws, lw0, lw1, lw2, lw3, dw0, dw1, dw2, dw3);

    fd_deformer_kernel<<<PI_/128, 128, SMEM_BYTES, stream>>>(
        ws, xd, cond, tfs,
        lb0, lb1, lb2, lb3, db0, db1, db2, db3,
        (float*)d_out);
}
